// GNNModel_33200097198381
// MI455X (gfx1250) — matrix operand order repaired and checked by reading, not compiled, not run on hardware
//
#include <hip/hip_runtime.h>
#include <hip/hip_bf16.h>

typedef __attribute__((ext_vector_type(16))) __bf16 v16bf;
typedef __attribute__((ext_vector_type(8)))  float  v8f;
typedef __attribute__((ext_vector_type(4)))  unsigned int v4u;
typedef __attribute__((ext_vector_type(8)))  int v8i;
typedef __attribute__((ext_vector_type(4)))  int v4i;

#define D_FEAT 768
#define N_HEAD 8
#define C_HEAD 96
#define N_LAYER 6
#define N_GRAPH 128
#define NEG_SLOPE 0.2f

#if defined(__has_builtin)
#if __has_builtin(__builtin_amdgcn_tensor_load_to_lds)
#define HAVE_TDM 1
#endif
#if __has_builtin(__builtin_amdgcn_s_wait_tensorcnt)
#define HAVE_TENSORCNT 1
#endif
#endif

__device__ __forceinline__ void wait_tensorcnt0() {
#ifdef HAVE_TENSORCNT
  __builtin_amdgcn_s_wait_tensorcnt(0);
#else
  asm volatile("s_wait_tensorcnt 0x0" ::: "memory");
#endif
}

// ---------------------------------------------------------------------------
// Embedding gather: x[i,:] = emb[nodes[i],:]; also emits bf16 shadow xb
// ---------------------------------------------------------------------------
__global__ __launch_bounds__(192) void k_gather_emb(const int* __restrict__ nodes,
                                                    const float* __restrict__ emb,
                                                    float* __restrict__ x,
                                                    __bf16* __restrict__ xb) {
  int node = blockIdx.x;
  int t = threadIdx.x;                       // 0..191, float4 -> 768 floats
  float4 v = ((const float4*)(emb + (size_t)nodes[node] * D_FEAT))[t];
  ((float4*)(x + (size_t)node * D_FEAT))[t] = v;
  union { __bf16 b[4]; uint2 u; } pk;
  pk.b[0] = (__bf16)v.x; pk.b[1] = (__bf16)v.y;
  pk.b[2] = (__bf16)v.z; pk.b[3] = (__bf16)v.w;
  *(uint2*)(xb + (size_t)node * D_FEAT + t * 4) = pk.u;
}

// ---------------------------------------------------------------------------
// fp32 -> bf16 (weights, all layers at once)
// ---------------------------------------------------------------------------
__global__ __launch_bounds__(256) void k_f32_to_bf16(const float* __restrict__ in,
                                                     __bf16* __restrict__ out, long n) {
  long i = (long)blockIdx.x * blockDim.x + threadIdx.x;
  if (i >= n) return;
  out[i] = (__bf16)in[i];
}

// ---------------------------------------------------------------------------
// Tiled bf16 WMMA GEMM: C[M,N] = A[M,K] * B[K,N]
// Block tile 128(M) x 64(N), K-step 32, 256 threads = 8 waves.
// A tile staged via Tensor Data Mover (async DMA to LDS, TENSORcnt-fenced);
// B tile staged as [n][k] so each lane's fragment load is 2x ds_load_b128.
// ---------------------------------------------------------------------------
#define TM 128
#define TN 64
#define TK 32

union FragBF { v16bf v; unsigned int u[8]; uint4 q[2]; };
union PackB  { __bf16 b[8]; uint4 q; };

__global__ __launch_bounds__(256) void k_gemm_bf16(const __bf16* __restrict__ A,
                                                   const __bf16* __restrict__ B,
                                                   float* __restrict__ C,
                                                   int M, int N, int K) {
  __shared__ __bf16 As[TM * TK];   // row-major [m][k]  (TDM-written)
  __shared__ __bf16 Bs[TN * TK];   // transposed [n][k]

  const int tid  = threadIdx.x;
  const int wave = tid >> 5;
  const int lane = tid & 31;
  const int hi   = lane >> 4;      // lane half (0/1)
  const int lr   = lane & 15;

  const int m0 = blockIdx.x * TM;
  const int n0 = blockIdx.y * TN;

  v8f acc[4] = {};

  for (int k0 = 0; k0 < K; k0 += TK) {
#ifdef HAVE_TDM
    // ---- TDM: DMA the 128x32 bf16 A tile into LDS (issued once, wave 0) ----
    if (tid == 0) {
      unsigned lds = (unsigned)(size_t)(void*)As;
      unsigned long long ga =
          (unsigned long long)(size_t)(A + (size_t)m0 * K + k0);
      unsigned td0 = (unsigned)(K - k0);   // remaining extent dim0 (elements)
      unsigned td1 = (unsigned)(M - m0);   // remaining rows -> OOB rows read 0
      v4u g0;
      g0.x = 1u;                                        // count=1 valid D#
      g0.y = lds;                                       // lds_addr (bytes)
      g0.z = (unsigned)(ga & 0xFFFFFFFFu);              // global_addr lo
      g0.w = (unsigned)((ga >> 32) & 0x1FFFFFFu) | (2u << 30);  // hi | type=2
      v8i g1;
      g1[0] = (int)(1u << 16);                          // data_size=1 (2 bytes)
      g1[1] = (int)((td0 & 0xFFFFu) << 16);             // tensor_dim0[15:0]
      g1[2] = (int)(((td0 >> 16) & 0xFFFFu) | ((td1 & 0xFFFFu) << 16));
      g1[3] = (int)(((td1 >> 16) & 0xFFFFu) | ((unsigned)TK << 16)); // tile_dim0
      g1[4] = (int)(unsigned)TM;                        // tile_dim1
      g1[5] = (int)(unsigned)K;                         // tensor_dim0_stride lo
      g1[6] = 0;
      g1[7] = 0;
      v4i z4 = {0, 0, 0, 0};
      v8i z8 = {0, 0, 0, 0, 0, 0, 0, 0};
      __builtin_amdgcn_tensor_load_to_lds(g0, g1, z4, z4, z8, 0);
    }
#else
    // ---- fallback: VALU staging of A tile ----
    {
      int r = tid >> 1;
      int c = (tid & 1) * 16;
      int gr = m0 + r;
      uint4 z0 = make_uint4(0u, 0u, 0u, 0u), z1 = z0;
      if (gr < M) {
        const uint4* p = (const uint4*)(A + (size_t)gr * K + k0 + c);
        z0 = p[0];
        z1 = p[1];
      }
      uint4* q = (uint4*)(As + r * TK + c);
      q[0] = z0;
      q[1] = z1;
    }
#endif
    // ---- stage B tile transposed: thread packs 8 K-values, one b128 store ----
    {
      int n  = tid & 63;            // column within tile
      int kb = (tid >> 6) * 8;      // 0,8,16,24
      const __bf16* p = B + (size_t)(k0 + kb) * N + n0 + n;
      PackB pk;
#pragma unroll
      for (int j = 0; j < 8; ++j) pk.b[j] = p[(size_t)j * N];
      *(uint4*)(Bs + n * TK + kb) = pk.q;
    }
#ifdef HAVE_TDM
    if (tid == 0) wait_tensorcnt0();
#endif
    __syncthreads();

    // ---- A fragment (16x32): lane=row, VGPR v holds K = hi*16 + 2*v ----
    FragBF a;
#pragma unroll
    for (int v = 0; v < 8; ++v) {
      int kk = hi * 16 + 2 * v;
      a.u[v] = *(const unsigned int*)(As + (wave * 16 + lr) * TK + kk);
    }
    // ---- B fragments (32x16): lane=col, lane-half selects K 0-15 / 16-31 ----
#pragma unroll
    for (int t = 0; t < 4; ++t) {
      FragBF b;
      const uint4* p = (const uint4*)(Bs + (t * 16 + lr) * TK + hi * 16);
      b.q[0] = p[0];
      b.q[1] = p[1];
      acc[t] = __builtin_amdgcn_wmma_f32_16x16x32_bf16(
          false, a.v, false, b.v, (short)0, acc[t], false, false);
    }
    __syncthreads();
  }

  // ---- store C: VGPR i -> row (i + hi*8), col = lr ----
#pragma unroll
  for (int t = 0; t < 4; ++t) {
#pragma unroll
    for (int i = 0; i < 8; ++i) {
      int row = m0 + wave * 16 + hi * 8 + i;
      int col = n0 + t * 16 + lr;
      if (row < M) C[(size_t)row * N + col] = acc[t][i];
    }
  }
}

// ---------------------------------------------------------------------------
// alpha_s / alpha_d: per (node, head) dot products of length 96
// ---------------------------------------------------------------------------
__global__ __launch_bounds__(256) void k_alpha(const float* __restrict__ h,
                                               const float* __restrict__ a_s,
                                               const float* __restrict__ a_d,
                                               float* __restrict__ as_o,
                                               float* __restrict__ ad_o,
                                               int nn) {
  int idx = blockIdx.x * blockDim.x + threadIdx.x;
  if (idx >= nn * N_HEAD) return;
  int node = idx >> 3, hd = idx & 7;
  const float* hp = h + (size_t)node * D_FEAT + hd * C_HEAD;
  const float* ws = a_s + hd * C_HEAD;
  const float* wd = a_d + hd * C_HEAD;
  float ss = 0.0f, sd = 0.0f;
#pragma unroll 4
  for (int c = 0; c < C_HEAD; ++c) {
    float v = hp[c];
    ss += v * ws[c];
    sd += v * wd[c];
  }
  as_o[idx] = ss;
  ad_o[idx] = sd;
}

// ---------------------------------------------------------------------------
// CSR-by-destination build (degree count -> scan -> fill)
// ---------------------------------------------------------------------------
__global__ __launch_bounds__(256) void k_zero_int(int* __restrict__ a, int n) {
  int i = blockIdx.x * blockDim.x + threadIdx.x;
  if (i < n) a[i] = 0;
}

__global__ __launch_bounds__(256) void k_deg_count(const int* __restrict__ ei,
                                                   int ne, int nn,
                                                   int* __restrict__ deg) {
  int e = blockIdx.x * blockDim.x + threadIdx.x;
  if (e >= ne + nn) return;
  int d = (e < ne) ? ei[ne + e] : (e - ne);
  atomicAdd(&deg[d], 1);
}

__global__ __launch_bounds__(1024) void k_scan(const int* __restrict__ deg,
                                               int* __restrict__ rowptr, int nn) {
  __shared__ int buf[1024];
  __shared__ int carry;
  if (threadIdx.x == 0) carry = 0;
  __syncthreads();
  for (int base = 0; base < nn; base += 1024) {
    int i = base + threadIdx.x;
    int v = (i < nn) ? deg[i] : 0;
    buf[threadIdx.x] = v;
    __syncthreads();
    for (int ofs = 1; ofs < 1024; ofs <<= 1) {
      int t = (threadIdx.x >= ofs) ? buf[threadIdx.x - ofs] : 0;
      __syncthreads();
      buf[threadIdx.x] += t;
      __syncthreads();
    }
    int inc = buf[threadIdx.x];
    if (i < nn) rowptr[i] = carry + inc - v;   // exclusive prefix
    __syncthreads();
    if (threadIdx.x == 1023) carry += buf[1023];
    __syncthreads();
  }
  if (threadIdx.x == 0) rowptr[nn] = carry;
}

__global__ __launch_bounds__(256) void k_csr_fill(const int* __restrict__ ei,
                                                  int ne, int nn,
                                                  const int* __restrict__ rowptr,
                                                  int* __restrict__ cursor,
                                                  int* __restrict__ csr_src) {
  int e = blockIdx.x * blockDim.x + threadIdx.x;
  if (e >= ne + nn) return;
  int s, d;
  if (e < ne) { s = ei[e]; d = ei[ne + e]; } else { s = d = e - ne; }
  int pos = atomicAdd(&cursor[d], 1);
  csr_src[rowptr[d] + pos] = s;
}

// ---------------------------------------------------------------------------
// per-(dst,head) edge softmax over CSR list -> alpha per slot (no atomics)
// ---------------------------------------------------------------------------
__device__ __forceinline__ float leaky(float v) {
  return v > 0.0f ? v : NEG_SLOPE * v;
}

__global__ __launch_bounds__(256) void k_softmax_csr(const int* __restrict__ rowptr,
                                                     const int* __restrict__ csr_src,
                                                     const float* __restrict__ as_o,
                                                     const float* __restrict__ ad_o,
                                                     float* __restrict__ alpha,
                                                     int nn) {
  int idx = blockIdx.x * blockDim.x + threadIdx.x;
  if (idx >= nn * N_HEAD) return;
  int d = idx >> 3, hd = idx & 7;
  int beg = rowptr[d], end = rowptr[d + 1];  // self-loop => non-empty
  float ad = ad_o[d * N_HEAD + hd];
  float mx = -3.402823466e38f;
  for (int j = beg; j < end; ++j) {
    float lg = leaky(as_o[csr_src[j] * N_HEAD + hd] + ad);
    mx = fmaxf(mx, lg);
  }
  float sum = 0.0f;
  for (int j = beg; j < end; ++j) {
    float e = __expf(leaky(as_o[csr_src[j] * N_HEAD + hd] + ad) - mx);
    alpha[(size_t)j * N_HEAD + hd] = e;
    sum += e;
  }
  float inv = 1.0f / sum;
  for (int j = beg; j < end; ++j) alpha[(size_t)j * N_HEAD + hd] *= inv;
}

// ---------------------------------------------------------------------------
// aggregation: one block per dst row, pure gather (no atomics).
// Emits fp32 x (for pooling) and relu'd bf16 xb (next layer's GEMM input).
// ---------------------------------------------------------------------------
__global__ __launch_bounds__(192) void k_message_csr(const int* __restrict__ rowptr,
                                                     const int* __restrict__ csr_src,
                                                     const float* __restrict__ h,
                                                     const float* __restrict__ alpha,
                                                     const float* __restrict__ bias,
                                                     float* __restrict__ x,
                                                     __bf16* __restrict__ xb) {
  int d = blockIdx.x;
  int t = threadIdx.x;                 // 0..191 float4; 96 % 4 == 0 -> no head split
  int hd = t / 24;                     // 24 float4 chunks per head
  float4 acc = ((const float4*)bias)[t];
  int beg = rowptr[d], end = rowptr[d + 1];
  for (int j = beg; j < end; ++j) {
    int s = csr_src[j];
    float al = alpha[(size_t)j * N_HEAD + hd];
    float4 v = ((const float4*)(h + (size_t)s * D_FEAT))[t];
    acc.x += al * v.x;
    acc.y += al * v.y;
    acc.z += al * v.z;
    acc.w += al * v.w;
  }
  ((float4*)(x + (size_t)d * D_FEAT))[t] = acc;
  union { __bf16 b[4]; uint2 u; } pk;
  pk.b[0] = (__bf16)fmaxf(acc.x, 0.0f);
  pk.b[1] = (__bf16)fmaxf(acc.y, 0.0f);
  pk.b[2] = (__bf16)fmaxf(acc.z, 0.0f);
  pk.b[3] = (__bf16)fmaxf(acc.w, 0.0f);
  *(uint2*)(xb + (size_t)d * D_FEAT + t * 4) = pk.u;
}

// ---------------------------------------------------------------------------
// ragged mean pooling
// ---------------------------------------------------------------------------
__global__ __launch_bounds__(256) void k_pool_zero(float* __restrict__ pool,
                                                   float* __restrict__ counts) {
  int i = blockIdx.x * blockDim.x + threadIdx.x;
  if (i < N_GRAPH * D_FEAT) pool[i] = 0.0f;
  if (i < N_GRAPH) counts[i] = 0.0f;
}

__global__ __launch_bounds__(192) void k_pool_acc(const int* __restrict__ seg,
                                                  const float* __restrict__ x,
                                                  float* __restrict__ pool,
                                                  float* __restrict__ counts) {
  int node = blockIdx.x;
  int g = seg[node];
  int t = threadIdx.x;
  float4 v = ((const float4*)(x + (size_t)node * D_FEAT))[t];
  float* p = pool + (size_t)g * D_FEAT + t * 4;
  atomicAdd(p + 0, v.x);
  atomicAdd(p + 1, v.y);
  atomicAdd(p + 2, v.z);
  atomicAdd(p + 3, v.w);
  if (t == 0) atomicAdd(&counts[g], 1.0f);
}

__global__ __launch_bounds__(192) void k_pool_div(const float* __restrict__ pool,
                                                  const float* __restrict__ counts,
                                                  float* __restrict__ out) {
  int g = blockIdx.x;
  int t = threadIdx.x;
  float c = counts[g];
  c = c > 1.0f ? c : 1.0f;
  float4 v = ((const float4*)(pool + (size_t)g * D_FEAT))[t];
  v.x /= c; v.y /= c; v.z /= c; v.w /= c;
  ((float4*)(out + (size_t)g * D_FEAT))[t] = v;
}

// ---------------------------------------------------------------------------
// host-side orchestration
// ---------------------------------------------------------------------------
extern "C" void kernel_launch(void* const* d_in, const int* in_sizes, int n_in,
                              void* d_out, int out_size, void* d_ws, size_t ws_size,
                              hipStream_t stream) {
  const int*   nodes   = (const int*)d_in[0];
  const int*   ei      = (const int*)d_in[1];   // [2, NE] flat: src row then dst row
  const int*   seg     = (const int*)d_in[2];
  const float* emb     = (const float*)d_in[4];
  const float* W       = (const float*)d_in[5]; // [L, D, D]
  const float* att_src = (const float*)d_in[6]; // [L, H, C]
  const float* att_dst = (const float*)d_in[7];
  const float* bias    = (const float*)d_in[8]; // [L, D]

  const int NN = in_sizes[0];
  const int NE = in_sizes[1] / 2;
  const int ET = NE + NN;                        // edges incl. self-loops

  // ---- carve workspace (256B aligned) ----
  size_t off = 0;
  auto carve = [&](size_t bytes) -> void* {
    void* p = (char*)d_ws + off;
    off += (bytes + 255) & ~(size_t)255;
    return p;
  };
  float*  x       = (float*)carve((size_t)NN * D_FEAT * 4);
  float*  h       = (float*)carve((size_t)NN * D_FEAT * 4);
  __bf16* xb      = (__bf16*)carve((size_t)NN * D_FEAT * 2);
  __bf16* wb      = (__bf16*)carve((size_t)N_LAYER * D_FEAT * D_FEAT * 2);
  float*  as_o    = (float*)carve((size_t)NN * N_HEAD * 4);
  float*  ad_o    = (float*)carve((size_t)NN * N_HEAD * 4);
  float*  alpha   = (float*)carve((size_t)ET * N_HEAD * 4);
  int*    deg     = (int*)carve((size_t)(NN + 1) * 4);
  int*    cursor  = (int*)carve((size_t)(NN + 1) * 4);
  int*    rowptr  = (int*)carve((size_t)(NN + 1) * 4);
  int*    csr_src = (int*)carve((size_t)ET * 4);
  float*  pool    = (float*)carve((size_t)N_GRAPH * D_FEAT * 4);
  float*  counts  = (float*)carve((size_t)N_GRAPH * 4);

  const long nw_all = (long)N_LAYER * D_FEAT * D_FEAT;

  // ---- one-time per call: gather, weight conversion, CSR build ----
  k_gather_emb<<<NN, 192, 0, stream>>>(nodes, emb, x, xb);
  k_f32_to_bf16<<<(int)((nw_all + 255) / 256), 256, 0, stream>>>(W, wb, nw_all);

  k_zero_int<<<(NN + 256) / 256, 256, 0, stream>>>(deg, NN + 1);
  k_zero_int<<<(NN + 256) / 256, 256, 0, stream>>>(cursor, NN + 1);
  k_deg_count<<<(ET + 255) / 256, 256, 0, stream>>>(ei, NE, NN, deg);
  k_scan<<<1, 1024, 0, stream>>>(deg, rowptr, NN);
  k_csr_fill<<<(ET + 255) / 256, 256, 0, stream>>>(ei, NE, NN, rowptr, cursor, csr_src);

  // ---- layers ----
  for (int l = 0; l < N_LAYER; ++l) {
    dim3 ggrid((NN + TM - 1) / TM, D_FEAT / TN);
    k_gemm_bf16<<<ggrid, 256, 0, stream>>>(
        xb, wb + (size_t)l * D_FEAT * D_FEAT, h, NN, D_FEAT, D_FEAT);

    k_alpha<<<(NN * N_HEAD + 255) / 256, 256, 0, stream>>>(
        h, att_src + (size_t)l * N_HEAD * C_HEAD,
        att_dst + (size_t)l * N_HEAD * C_HEAD, as_o, ad_o, NN);

    k_softmax_csr<<<(NN * N_HEAD + 255) / 256, 256, 0, stream>>>(
        rowptr, csr_src, as_o, ad_o, alpha, NN);

    k_message_csr<<<NN, 192, 0, stream>>>(
        rowptr, csr_src, h, alpha, bias + (size_t)l * D_FEAT, x, xb);
  }

  // ---- per-graph mean pooling ----
  k_pool_zero<<<(N_GRAPH * D_FEAT + 255) / 256, 256, 0, stream>>>(pool, counts);
  k_pool_acc<<<NN, 192, 0, stream>>>(seg, x, pool, counts);
  k_pool_div<<<N_GRAPH, 192, 0, stream>>>(pool, counts, (float*)d_out);
}
